// GATClassifier_18657337934554
// MI455X (gfx1250) — matrix operand order repaired and checked by reading, not compiled, not run on hardware
//
#include <hip/hip_runtime.h>
#include <cstddef>
#include <cstdint>

// ---------------------------------------------------------------------------
// Types for CDNA5 WMMA (wave32): v_wmma_f32_16x16x32_bf16
// ---------------------------------------------------------------------------
typedef __bf16 bf16;
typedef __bf16 v8bf  __attribute__((ext_vector_type(8)));
typedef __bf16 v16bf __attribute__((ext_vector_type(16)));
typedef float  v8f   __attribute__((ext_vector_type(8)));
// Parameter type of __builtin_amdgcn_global_load_async_to_lds_b128 (from the
// compiler diagnostic): pointer to 16-byte int vector, generic AS, non-const.
typedef int b128_t __attribute__((vector_size(16)));

#if __has_builtin(__builtin_amdgcn_global_load_async_to_lds_b128)
#define USE_ASYNC_LDS 1
#else
#define USE_ASYNC_LDS 0
#endif

static __device__ __forceinline__ void async_wait0() {
#if __has_builtin(__builtin_amdgcn_s_wait_asynccnt)
  __builtin_amdgcn_s_wait_asynccnt(0);
#else
  asm volatile("s_wait_asynccnt 0" ::: "memory");
#endif
}

static __device__ __forceinline__ bf16 f2bf(float f) {
  // round-to-nearest-even fp32 -> bf16
  unsigned u = __builtin_bit_cast(unsigned, f);
  u += 0x7FFFu + ((u >> 16) & 1u);
  unsigned short h = (unsigned short)(u >> 16);
  return __builtin_bit_cast(bf16, h);
}

// ---------------------------------------------------------------------------
// Elementwise cast fp32 -> bf16
// ---------------------------------------------------------------------------
__global__ void __launch_bounds__(256) cast_bf_kernel(const float* __restrict__ src,
                                                      bf16* __restrict__ dst, size_t n) {
  size_t i = (size_t)blockIdx.x * 256 + threadIdx.x;
  if (i < n) dst[i] = f2bf(src[i]);
}

// ---------------------------------------------------------------------------
// rnorm[n] = 1/||df[n,:]||  (one block per row)
// ---------------------------------------------------------------------------
__global__ void __launch_bounds__(256) rnorm_kernel(const float* __restrict__ df,
                                                    float* __restrict__ rn, int K) {
  __shared__ float red[256];
  const int n = blockIdx.x, tid = threadIdx.x;
  const float* row = df + (size_t)n * K;
  float s = 0.f;
  for (int k = tid; k < K; k += 256) { float v = row[k]; s += v * v; }
  red[tid] = s; __syncthreads();
  for (int off = 128; off > 0; off >>= 1) {
    if (tid < off) red[tid] += red[tid + off];
    __syncthreads();
  }
  if (tid == 0) rn[n] = rsqrtf(red[0]);
}

// ---------------------------------------------------------------------------
// In-place: S (= df@df^T raw dot) -> log edge multiplicity.
// mult = (cos>0) + (i==j) in {0,1,2} -> logm in {-inf, 0, ln2}. Ns = 4096.
// ---------------------------------------------------------------------------
__global__ void __launch_bounds__(256) logm_kernel(float* __restrict__ S,
                                                   const float* __restrict__ rn) {
  size_t idx = (size_t)blockIdx.x * 256 + threadIdx.x;
  int i = (int)(idx >> 12);
  int j = (int)(idx & 4095);
  float c = S[idx] * rn[i] * rn[j];
  int mult = (c > 0.f ? 1 : 0) + (i == j ? 1 : 0);
  S[idx] = (mult == 0) ? -__builtin_inff() : (mult == 1 ? 0.f : 0.6931471805599453f);
}

// ---------------------------------------------------------------------------
// Transpose + cast: src fp32 [R,C] -> dst bf16 [C,R]
// ---------------------------------------------------------------------------
__global__ void __launch_bounds__(256) tcast_kernel(const float* __restrict__ src,
                                                    bf16* __restrict__ dst, int R, int C) {
  __shared__ float tile[32][33];
  const int tx = threadIdx.x & 31, ty = threadIdx.x >> 5;  // 32 x 8
  const int x = blockIdx.x * 32 + tx;
  const int y0 = blockIdx.y * 32;
#pragma unroll
  for (int j = 0; j < 32; j += 8) {
    int y = y0 + ty + j;
    if (x < C && y < R) tile[ty + j][tx] = src[(size_t)y * C + x];
  }
  __syncthreads();
  const int xo = blockIdx.y * 32 + tx;   // col of dst (= row of src)
  const int yo0 = blockIdx.x * 32;       // row of dst (= col of src)
#pragma unroll
  for (int j = 0; j < 32; j += 8) {
    int yo = yo0 + ty + j;
    if (xo < R && yo < C) dst[(size_t)yo * R + xo] = f2bf(tile[tx][ty + j]);
  }
}

// ---------------------------------------------------------------------------
// bf16 WMMA GEMM: C[M,N] = A[M,K] * BT[N,K]^T   (fp32 accumulate)
// Block tile 128x128, BK=32, 256 threads (8 waves), wave tile 32x64
// (2 x 4 subtiles of 16x16 -> 8 v_wmma_f32_16x16x32_bf16 per K-step).
// Double-buffered LDS; staging via GLOBAL_LOAD_ASYNC_TO_LDS_B128 (ASYNCcnt)
// when the toolchain exposes it, else a register-pipelined prefetch.
// Requires: M,N multiples of 128; K multiple of 32. Epilogue:
//   epil 0 = none, 1 = relu, 2 = leaky(0.01); bias added if non-null;
//   accum != 0 -> C += result.
// ---------------------------------------------------------------------------
__global__ void __launch_bounds__(256) gemm_bf16_wmma(
    const bf16* __restrict__ A, const bf16* __restrict__ BT, float* __restrict__ C,
    int M, int N, int K, const float* __restrict__ bias, int epil, int accum) {
  (void)M;
  __shared__ bf16 As[2][128 * 40];   // [buf][row*40 + k]; stride 40 => 80 B rows, 16-B aligned
  __shared__ bf16 Bs[2][128 * 40];   // [buf][col*40 + k]
  const int tid = threadIdx.x;
  const int lane = tid & 31, wid = tid >> 5;
  const int l16 = lane & 15, lhalf = lane >> 4;
  const int wm = (wid & 3) * 32;      // wave row offset inside block tile
  const int wn = (wid >> 2) * 64;     // wave col offset inside block tile
  const int bm = blockIdx.y * 128;
  const int bn = blockIdx.x * 128;

  // This thread's two 16-byte staging chunks (512 chunks / 256 threads).
  const int r0 = tid >> 2, p0 = tid & 3;
  const int r1 = (tid + 256) >> 2, p1 = (tid + 256) & 3;

  v8f acc[2][4] = {};
  const int nk = K >> 5;

#if USE_ASYNC_LDS
  auto issue_tile = [&](int k0, int buf) {
    __builtin_amdgcn_global_load_async_to_lds_b128(
        (b128_t*)&A[(size_t)(bm + r0) * K + k0 + p0 * 8],
        (b128_t*)&As[buf][r0 * 40 + p0 * 8], 0, 0);
    __builtin_amdgcn_global_load_async_to_lds_b128(
        (b128_t*)&BT[(size_t)(bn + r0) * K + k0 + p0 * 8],
        (b128_t*)&Bs[buf][r0 * 40 + p0 * 8], 0, 0);
    __builtin_amdgcn_global_load_async_to_lds_b128(
        (b128_t*)&A[(size_t)(bm + r1) * K + k0 + p1 * 8],
        (b128_t*)&As[buf][r1 * 40 + p1 * 8], 0, 0);
    __builtin_amdgcn_global_load_async_to_lds_b128(
        (b128_t*)&BT[(size_t)(bn + r1) * K + k0 + p1 * 8],
        (b128_t*)&Bs[buf][r1 * 40 + p1 * 8], 0, 0);
  };
#else
  uint4 ra0, rb0, ra1, rb1;
  auto load_tile = [&](int k0) {
    ra0 = *reinterpret_cast<const uint4*>(&A[(size_t)(bm + r0) * K + k0 + p0 * 8]);
    rb0 = *reinterpret_cast<const uint4*>(&BT[(size_t)(bn + r0) * K + k0 + p0 * 8]);
    ra1 = *reinterpret_cast<const uint4*>(&A[(size_t)(bm + r1) * K + k0 + p1 * 8]);
    rb1 = *reinterpret_cast<const uint4*>(&BT[(size_t)(bn + r1) * K + k0 + p1 * 8]);
  };
  auto store_tile = [&](int buf) {
    *reinterpret_cast<uint4*>(&As[buf][r0 * 40 + p0 * 8]) = ra0;
    *reinterpret_cast<uint4*>(&Bs[buf][r0 * 40 + p0 * 8]) = rb0;
    *reinterpret_cast<uint4*>(&As[buf][r1 * 40 + p1 * 8]) = ra1;
    *reinterpret_cast<uint4*>(&Bs[buf][r1 * 40 + p1 * 8]) = rb1;
  };
#endif

  auto compute = [&](int buf) {
    // A fragment (16x32): lane half 0 -> K 0..7 & 16..23; half 1 -> 8..15 & 24..31
    v16bf af[2];
#pragma unroll
    for (int i = 0; i < 2; ++i) {
      const bf16* pa = &As[buf][(wm + i * 16 + l16) * 40 + lhalf * 8];
      v8bf lo = *reinterpret_cast<const v8bf*>(pa);
      v8bf hi = *reinterpret_cast<const v8bf*>(pa + 16);
      af[i] = __builtin_shufflevector(lo, hi, 0, 1, 2, 3, 4, 5, 6, 7,
                                      8, 9, 10, 11, 12, 13, 14, 15);
    }
    // B fragment (32x16): lane half 0 -> K 0..7 & 16..23; half 1 -> 8..15 & 24..31
    v16bf bfm[4];
#pragma unroll
    for (int j = 0; j < 4; ++j) {
      const bf16* pb = &Bs[buf][(wn + j * 16 + l16) * 40 + lhalf * 8];
      v8bf lo = *reinterpret_cast<const v8bf*>(pb);
      v8bf hi = *reinterpret_cast<const v8bf*>(pb + 16);
      bfm[j] = __builtin_shufflevector(lo, hi, 0, 1, 2, 3, 4, 5, 6, 7,
                                      8, 9, 10, 11, 12, 13, 14, 15);
    }
#pragma unroll
    for (int i = 0; i < 2; ++i)
#pragma unroll
      for (int j = 0; j < 4; ++j)
        acc[i][j] = __builtin_amdgcn_wmma_f32_16x16x32_bf16(
            false, af[i], false, bfm[j], (short)0, acc[i][j], false, false);
  };

#if USE_ASYNC_LDS
  // Async double-buffer pipeline: wait own ASYNCcnt, barrier (tile complete for
  // all waves), issue next tile into the other buffer, compute current.
  issue_tile(0, 0);
  for (int k = 0; k < nk; ++k) {
    async_wait0();
    __syncthreads();
    if (k + 1 < nk) issue_tile((k + 1) << 5, (k + 1) & 1);
    compute(k & 1);
  }
#else
  // Register-pipelined double buffer: one barrier/iter; next tile's global
  // loads in flight while WMMAs run; store to alternate buffer after compute.
  load_tile(0);
  store_tile(0);
  for (int k = 0; k < nk; ++k) {
    __syncthreads();
    if (k + 1 < nk) load_tile((k + 1) << 5);
    compute(k & 1);
    if (k + 1 < nk) store_tile((k + 1) & 1);
  }
#endif

  // Epilogue. C/D layout: VGPR r -> row (r + 8*half), col = lane&15 of the tile.
#pragma unroll
  for (int i = 0; i < 2; ++i) {
    const int row0 = bm + wm + i * 16 + lhalf * 8;
#pragma unroll
    for (int j = 0; j < 4; ++j) {
      const int col = bn + wn + j * 16 + l16;
      const float bv = bias ? bias[col] : 0.f;
#pragma unroll
      for (int r = 0; r < 8; ++r) {
        size_t idx = (size_t)(row0 + r) * N + col;
        float v = acc[i][j][r] + bv;
        if (accum) v += C[idx];
        if (epil == 1) v = fmaxf(v, 0.f);
        else if (epil == 2) v = v > 0.f ? v : 0.01f * v;
        C[idx] = v;
      }
    }
  }
}

// ---------------------------------------------------------------------------
// el[h,n] = feat[n,h,:] . a_l[h,:] ; er likewise. Grid (N, H).
// ---------------------------------------------------------------------------
__global__ void __launch_bounds__(256) elr_kernel(
    const float* __restrict__ feat, const float* __restrict__ al,
    const float* __restrict__ ar, float* __restrict__ el, float* __restrict__ er,
    int H, int D, int N) {
  __shared__ float redl[256], redr[256];
  const int n = blockIdx.x, h = blockIdx.y, tid = threadIdx.x;
  const float* f  = feat + (size_t)n * H * D + (size_t)h * D;
  const float* wl = al + (size_t)h * D;
  const float* wr = ar + (size_t)h * D;
  float sl = 0.f, sr = 0.f;
  for (int e = tid; e < D; e += 256) { float v = f[e]; sl += v * wl[e]; sr += v * wr[e]; }
  redl[tid] = sl; redr[tid] = sr; __syncthreads();
  for (int off = 128; off > 0; off >>= 1) {
    if (tid < off) { redl[tid] += redl[tid + off]; redr[tid] += redr[tid + off]; }
    __syncthreads();
  }
  if (tid == 0) { el[(size_t)h * N + n] = redl[0]; er[(size_t)h * N + n] = redr[0]; }
}

// ---------------------------------------------------------------------------
// Edge softmax (logm symmetric -> softmax over sources = row softmax of logm).
// One block per destination row d; writes alphaT[d, :] in bf16 (A of agg GEMM).
// ---------------------------------------------------------------------------
__global__ void __launch_bounds__(256) attn_softmax_kernel(
    const float* __restrict__ logm, const float* __restrict__ el_h,
    const float* __restrict__ er_h, bf16* __restrict__ alphaT, int Ns) {
  __shared__ float rowbuf[4096];
  __shared__ float red[256];
  const int d = blockIdx.x, tid = threadIdx.x;
  const float* lrow = logm + (size_t)d * Ns;
  const float erd = er_h[d];
  float lmax = -__builtin_inff();
  for (int s = tid; s < Ns; s += 256) {
    float e = el_h[s] + erd;
    e = e > 0.f ? e : 0.2f * e;          // leaky_relu(., 0.2)
    float v = e + lrow[s];               // + log multiplicity mask
    rowbuf[s] = v;
    lmax = fmaxf(lmax, v);
  }
  red[tid] = lmax; __syncthreads();
  for (int off = 128; off > 0; off >>= 1) {
    if (tid < off) red[tid] = fmaxf(red[tid], red[tid + off]);
    __syncthreads();
  }
  const float m = red[0]; __syncthreads();
  float lsum = 0.f;
  for (int s = tid; s < Ns; s += 256) {
    float e = __expf(rowbuf[s] - m);     // exp(-inf - m) == 0 masks non-edges
    rowbuf[s] = e;
    lsum += e;
  }
  red[tid] = lsum; __syncthreads();
  for (int off = 128; off > 0; off >>= 1) {
    if (tid < off) red[tid] += red[tid + off];
    __syncthreads();
  }
  const float inv = 1.f / red[0];
  bf16* out = alphaT + (size_t)d * Ns;
  for (int s = tid; s < Ns; s += 256) out[s] = f2bf(rowbuf[s] * inv);
}

// ---------------------------------------------------------------------------
// xcat[n, 0:2048] = df ; xcat[n, 2048:2560] = out2 (bf16)
// ---------------------------------------------------------------------------
__global__ void __launch_bounds__(256) concat_kernel(
    const float* __restrict__ df, const float* __restrict__ out2,
    bf16* __restrict__ xcat, size_t total) {
  size_t idx = (size_t)blockIdx.x * 256 + threadIdx.x;
  if (idx >= total) return;
  int n = (int)(idx / 2560);
  int c = (int)(idx % 2560);
  float v = (c < 2048) ? df[(size_t)n * 2048 + c] : out2[(size_t)n * 512 + (c - 2048)];
  xcat[idx] = f2bf(v);
}

// ---------------------------------------------------------------------------
// Tiny classifier head: out[n,c] = b[c] + x[n,:] . W[:,c]   (C = 10, VALU)
// ---------------------------------------------------------------------------
__global__ void __launch_bounds__(256) classifier_kernel(
    const float* __restrict__ x, const float* __restrict__ W,
    const float* __restrict__ b, float* __restrict__ out, int N, int K, int C) {
  int idx = blockIdx.x * 256 + threadIdx.x;
  if (idx >= N * C) return;
  int n = idx / C, c = idx % C;
  const float* xr = x + (size_t)n * K;
  float s = b[c];
  for (int k = 0; k < K; ++k) s += xr[k] * W[(size_t)k * C + c];
  out[idx] = s;
}

// ---------------------------------------------------------------------------
// Host-side orchestration
// ---------------------------------------------------------------------------
extern "C" void kernel_launch(void* const* d_in, const int* in_sizes, int n_in,
                              void* d_out, int out_size, void* d_ws, size_t ws_size,
                              hipStream_t stream) {
  (void)in_sizes; (void)n_in; (void)out_size; (void)ws_size;
  const int N = 4096, EMB = 2048, C = 10;
  const float* df    = (const float*)d_in[0];
  const float* W1    = (const float*)d_in[1];
  const float* al1   = (const float*)d_in[2];
  const float* ar1   = (const float*)d_in[3];
  const float* W2    = (const float*)d_in[4];
  const float* al2   = (const float*)d_in[5];
  const float* ar2   = (const float*)d_in[6];
  const float* lin1w = (const float*)d_in[7];
  const float* lin1b = (const float*)d_in[8];
  const float* c1w   = (const float*)d_in[9];
  const float* c1b   = (const float*)d_in[10];
  const float* c2w   = (const float*)d_in[11];
  const float* c2b   = (const float*)d_in[12];
  float* logits1 = (float*)d_out;
  float* logits2 = logits1 + (size_t)N * C;

  char* wp = (char*)d_ws;
  auto carve = [&](size_t bytes) -> char* {
    char* p = wp;
    wp += (bytes + 255) & ~(size_t)255;
    return p;
  };
  float* S      = (float*)carve((size_t)N * N * 4);        // sim -> logm (64 MB, L2-res)
  bf16*  alphaT = (bf16*) carve((size_t)N * N * 2);        // attention (reused per head)
  bf16*  df_bf  = (bf16*) carve((size_t)N * EMB * 2);
  float* featF  = (float*)carve((size_t)N * 1024 * 4);     // feat1 -> x1 -> feat2 -> xlin
  bf16*  featT  = (bf16*) carve((size_t)1024 * N * 2);     // feat^T bf16 (GEMM BT operand)
  bf16*  x1_bf  = (bf16*) carve((size_t)N * 1024 * 2);
  bf16*  wT     = (bf16*) carve((size_t)2560 * 1024 * 2);  // reused for W1^T/W2^T/lin1^T
  float* out2F  = (float*)carve((size_t)N * 512 * 4);
  bf16*  xcat   = (bf16*) carve((size_t)N * 2560 * 2);
  float* rn     = (float*)carve((size_t)N * 4);
  float* el1    = (float*)carve((size_t)N * 4);
  float* er1    = (float*)carve((size_t)N * 4);
  float* el2    = (float*)carve((size_t)2 * N * 4);
  float* er2    = (float*)carve((size_t)2 * N * 4);

  const dim3 b256(256);

  // --- graph construction (cosine threshold) ---
  cast_bf_kernel<<<(unsigned)((size_t)N * EMB / 256), b256, 0, stream>>>(df, df_bf, (size_t)N * EMB);
  rnorm_kernel<<<N, b256, 0, stream>>>(df, rn, EMB);
  gemm_bf16_wmma<<<dim3(N / 128, N / 128), b256, 0, stream>>>(df_bf, df_bf, S, N, N, EMB, nullptr, 0, 0);
  logm_kernel<<<(unsigned)((size_t)N * N / 256), b256, 0, stream>>>(S, rn);

  // --- GAT layer 1 (H=1, D=1024) ---
  tcast_kernel<<<dim3(1024 / 32, 2048 / 32), b256, 0, stream>>>(W1, wT, 2048, 1024);
  gemm_bf16_wmma<<<dim3(1024 / 128, N / 128), b256, 0, stream>>>(df_bf, wT, featF, N, 1024, EMB, nullptr, 0, 0);
  elr_kernel<<<dim3(N, 1), b256, 0, stream>>>(featF, al1, ar1, el1, er1, 1, 1024, N);
  tcast_kernel<<<dim3(1024 / 32, N / 32), b256, 0, stream>>>(featF, featT, N, 1024);
  attn_softmax_kernel<<<N, b256, 0, stream>>>(S, el1, er1, alphaT, N);
  gemm_bf16_wmma<<<dim3(1024 / 128, N / 128), b256, 0, stream>>>(alphaT, featT, featF, N, 1024, N, nullptr, 1, 0); // relu
  cast_bf_kernel<<<(unsigned)((size_t)N * 1024 / 256), b256, 0, stream>>>(featF, x1_bf, (size_t)N * 1024);

  // --- GAT layer 2 (H=2, D=512), heads summed ---
  tcast_kernel<<<dim3(1024 / 32, 1024 / 32), b256, 0, stream>>>(W2, wT, 1024, 1024);
  gemm_bf16_wmma<<<dim3(1024 / 128, N / 128), b256, 0, stream>>>(x1_bf, wT, featF, N, 1024, 1024, nullptr, 0, 0);
  elr_kernel<<<dim3(N, 2), b256, 0, stream>>>(featF, al2, ar2, el2, er2, 2, 512, N);
  tcast_kernel<<<dim3(1024 / 32, N / 32), b256, 0, stream>>>(featF, featT, N, 1024);
  for (int h = 0; h < 2; ++h) {
    attn_softmax_kernel<<<N, b256, 0, stream>>>(S, el2 + (size_t)h * N, er2 + (size_t)h * N, alphaT, N);
    gemm_bf16_wmma<<<dim3(512 / 128, N / 128), b256, 0, stream>>>(
        alphaT, featT + (size_t)h * 512 * N, out2F, N, 512, N, nullptr, 0, h);  // h==1 accumulates
  }

  // --- head: concat -> lin1 (leaky 0.01) -> classifiers ---
  concat_kernel<<<(unsigned)((size_t)N * 2560 / 256), b256, 0, stream>>>(df, out2F, xcat, (size_t)N * 2560);
  tcast_kernel<<<dim3(1024 / 32, 2560 / 32), b256, 0, stream>>>(lin1w, wT, 2560, 1024);
  gemm_bf16_wmma<<<dim3(1024 / 128, N / 128), b256, 0, stream>>>(xcat, wT, featF, N, 1024, 2560, lin1b, 2, 0);
  classifier_kernel<<<(N * C + 255) / 256, b256, 0, stream>>>(featF, c1w, c1b, logits1, N, 1024, C);
  classifier_kernel<<<(N * C + 255) / 256, b256, 0, stream>>>(df, c2w, c2b, logits2, N, EMB, C);
}
